// CDGS_45217415692540
// MI455X (gfx1250) — matrix operand order repaired and checked by reading, not compiled, not run on hardware
//
#include <hip/hip_runtime.h>
#include <hip/hip_bf16.h>
#include <math.h>

// ---------------------------------------------------------------------------
// CDGS layer for MI455X (gfx1250, wave32, WMMA).
// Heavy GEMMs -> v_wmma_f32_16x16x32_bf16; LDS tiles pre-swizzled into the
// WMMA lane layout (fragment loads = ds_load_b128).  Weights pre-converted to
// bf16 and staged with global_load_async_to_lds_b128 (ASYNCcnt path).
// GEMM is templated on the A-construction mode -> branch-free K loop.
// ---------------------------------------------------------------------------

typedef __attribute__((ext_vector_type(16))) __bf16 v16bf;
typedef __attribute__((ext_vector_type(8)))  __bf16 v8bf;
typedef __attribute__((ext_vector_type(8)))  float  v8f;

#define B_   8
#define N_   96
#define C_   256
#define H_   8
#define HC_  32
#define G_   32
#define NODES_    (B_*N_)          // 768
#define EDGEROWS_ (B_*N_*N_)       // 73728
#define EPS_ 1e-6f

__device__ __forceinline__ float siluf(float v) { return v / (1.f + __expf(-v)); }

// CDNA5 async global->LDS copy (16 bytes per lane), tracked by ASYNCcnt.
__device__ __forceinline__ void async_copy_b128(unsigned lds_off, const void* gptr)
{
    asm volatile("global_load_async_to_lds_b128 %0, %1, off"
                 :: "v"(lds_off), "v"((unsigned long long)(uintptr_t)gptr)
                 : "memory");
}
__device__ __forceinline__ void wait_async0()
{
    asm volatile("s_wait_asynccnt 0" ::: "memory");
}
// low 32 bits of a generic pointer into LDS == LDS byte offset (ISA 10.2)
__device__ __forceinline__ unsigned lds_off32(const void* p)
{
    return (unsigned)(uintptr_t)p;
}

// ===========================================================================
// WMMA GEMM:  out[m,n] = res[m,n]? + act( sum_k A(m,k)*W[n,k] + bias[n] )
//   MODE 0: A(m,k) = A[m*K+k]                                (f32)
//   MODE 1: A(m,k) = (A[m*K+k] + te[b*C+k]) * adj[m]         (h_edge rows, f32)
//   MODE 2: A(m,k) = hd[(b*N+i)*C+k] + hd[(b*N+j)*C+k]       (h_edge2 rows, f32)
//   MODE 3: A(m,k) = Abf[m*K+k]                              (bf16, async copy)
//   act: 0 none, 1 relu, 2 silu, 3 tanh
// Block: 128 threads = 4 waves; tile 64(M) x 64(N); K step 32.
// REQUIRES: M % 64 == 0, Nout % 64 == 0, K % 32 == 0 (true for all call sites).
// ===========================================================================
#define BM 64
#define BN 64
#define BK 32

__device__ __forceinline__ v8bf pack8(float a0, float a1, float a2, float a3,
                                      float a4, float a5, float a6, float a7)
{
    v8bf o;
    o[0] = (__bf16)a0; o[1] = (__bf16)a1; o[2] = (__bf16)a2; o[3] = (__bf16)a3;
    o[4] = (__bf16)a4; o[5] = (__bf16)a5; o[6] = (__bf16)a6; o[7] = (__bf16)a7;
    return o;
}

template <int MODE>
__global__ __launch_bounds__(128) void gemm_wmma_kernel(
    const float* __restrict__ A, const __bf16* __restrict__ Abf,
    const __bf16* __restrict__ Wbf,
    const float* __restrict__ bias, const float* __restrict__ res,
    float* __restrict__ out, __bf16* __restrict__ outbf,
    int K, int Nout, int act,
    const float* __restrict__ te, const float* __restrict__ adj,
    const float* __restrict__ hd)
{
    __shared__ v16bf As_f[4 * 32];   // 4 row-blocks x 32 lanes (4 KB)
    __shared__ v16bf Bs_f[4 * 32];   // 4 col-tiles  x 32 lanes (4 KB)

    const int tid  = threadIdx.x;
    const int wave = tid >> 5;
    const int lane = tid & 31;
    const int half = lane >> 4;
    const int l16  = lane & 15;
    const int m0 = blockIdx.x * BM;
    const int n0 = blockIdx.y * BN;

    v8f acc[4];
    #pragma unroll
    for (int t = 0; t < 4; ++t)
        #pragma unroll
        for (int r = 0; r < 8; ++r) acc[t][r] = 0.f;

    for (int k0 = 0; k0 < K; k0 += BK) {
        // ---- stage B tile via async copy of pre-converted bf16 weights ----
        #pragma unroll
        for (int it = 0; it < 2; ++it) {
            int task = tid + it * 128;          // 0..255
            int nn = task >> 2;                 // col in tile
            int qc = task & 3;                  // kk-chunk
            int kk0 = qc * 8;
            int t = nn >> 4, ln = nn & 15;
            int hf = qc & 1, eb = qc >> 1;
            unsigned loff = lds_off32((const char*)&Bs_f[t * 32 + hf * 16 + ln] + eb * 16);
            async_copy_b128(loff, Wbf + (long)(n0 + nn) * K + k0 + kk0);
        }
        // ---- stage A tile ----
        if (MODE == 3) {
            // pure bf16 copy, swizzle-preserving -> async
            #pragma unroll
            for (int it = 0; it < 2; ++it) {
                int task = tid + it * 128;
                int r = task >> 2, q = task & 3;
                int kc = k0 + q * 8;
                int wv = r >> 4, mm = r & 15;
                int hf = q & 1, eb = q >> 1;
                unsigned loff = lds_off32((const char*)&As_f[wv * 32 + hf * 16 + mm] + eb * 16);
                async_copy_b128(loff, Abf + (long)(m0 + r) * K + kc);
            }
        } else {
            #pragma unroll
            for (int it = 0; it < 2; ++it) {
                int task = tid + it * 128;          // 0..255
                int r = task >> 2;                  // row in tile
                int q = task & 3;                   // k-chunk (8 values)
                int m = m0 + r;
                int kc = k0 + q * 8;
                float4 a0, a1;
                if (MODE == 0) {
                    const float4* p = (const float4*)(A + (long)m * K + kc);
                    a0 = p[0]; a1 = p[1];
                    if (kc + BK < K) __builtin_prefetch(A + (long)m * K + kc + BK, 0, 1);
                } else if (MODE == 1) {
                    int b = m / (N_ * N_);
                    const float4* p = (const float4*)(A + (long)m * K + kc);
                    const float4* q4 = (const float4*)(te + b * C_ + kc);
                    float ad = adj[m];
                    float4 d0 = p[0], d1 = p[1], t0 = q4[0], t1 = q4[1];
                    a0.x = (d0.x + t0.x) * ad; a0.y = (d0.y + t0.y) * ad;
                    a0.z = (d0.z + t0.z) * ad; a0.w = (d0.w + t0.w) * ad;
                    a1.x = (d1.x + t1.x) * ad; a1.y = (d1.y + t1.y) * ad;
                    a1.z = (d1.z + t1.z) * ad; a1.w = (d1.w + t1.w) * ad;
                    if (kc + BK < K) __builtin_prefetch(A + (long)m * K + kc + BK, 0, 1);
                } else {
                    int b = m / (N_ * N_);
                    int rem = m - b * N_ * N_;
                    int i = rem / N_, j = rem - i * N_;
                    const float4* p = (const float4*)(hd + (long)(b * N_ + i) * C_ + kc);
                    const float4* q4 = (const float4*)(hd + (long)(b * N_ + j) * C_ + kc);
                    float4 d0 = p[0], d1 = p[1], t0 = q4[0], t1 = q4[1];
                    a0.x = d0.x + t0.x; a0.y = d0.y + t0.y;
                    a0.z = d0.z + t0.z; a0.w = d0.w + t0.w;
                    a1.x = d1.x + t1.x; a1.y = d1.y + t1.y;
                    a1.z = d1.z + t1.z; a1.w = d1.w + t1.w;
                }
                int wv = r >> 4, mm = r & 15;
                int hf = q & 1, eb = (q >> 1);
                ((v8bf*)&As_f[wv * 32 + hf * 16 + mm])[eb] =
                    pack8(a0.x, a0.y, a0.z, a0.w, a1.x, a1.y, a1.z, a1.w);
            }
        }
        wait_async0();
        __syncthreads();

        // ---- fragments are direct wide LDS loads ----
        v16bf af = As_f[wave * 32 + lane];
        #pragma unroll
        for (int t = 0; t < 4; ++t) {
            v16bf bfv = Bs_f[t * 32 + lane];
            acc[t] = __builtin_amdgcn_wmma_f32_16x16x32_bf16(
                false, af, false, bfv, (short)0, acc[t], false, false);
        }
        __syncthreads();
    }

    // ---- epilogue: bias + activation + residual, C/D 16x16 f32 layout ----
    #pragma unroll
    for (int t = 0; t < 4; ++t) {
        int n = n0 + t * 16 + l16;
        #pragma unroll
        for (int r = 0; r < 8; ++r) {
            int m = m0 + wave * 16 + half * 8 + r;
            float v = acc[t][r];
            if (bias) v += bias[n];
            if (act == 1)      v = fmaxf(v, 0.f);
            else if (act == 2) v = siluf(v);
            else if (act == 3) v = tanhf(v);
            if (res) v += res[(long)m * Nout + n];
            if (outbf) outbf[(long)m * Nout + n] = (__bf16)v;
            else       out[(long)m * Nout + n] = v;
        }
    }
}

// ===========================================================================
// Small kernels
// ===========================================================================

__global__ __launch_bounds__(256) void cvt_bf16_kernel(
    const float* __restrict__ s, __bf16* __restrict__ d, int n)
{
    int i = blockIdx.x * 256 + threadIdx.x;
    if (i < n) d[i] = (__bf16)s[i];
}

// tn/te = Linear(silu(temb))
__global__ __launch_bounds__(256) void temb_proj_kernel(
    const float* __restrict__ temb,
    const float* __restrict__ wn, const float* __restrict__ bn,
    const float* __restrict__ we, const float* __restrict__ be,
    float* __restrict__ tn, float* __restrict__ te)
{
    int idx = blockIdx.x * 256 + threadIdx.x;
    if (idx >= 2 * B_ * C_) return;
    int sel = idx >> 11;
    int r = idx & 2047;
    int b = r >> 8, n = r & 255;
    const float* W  = sel ? we : wn;
    const float* Bi = sel ? be : bn;
    float acc = 0.f;
    for (int c = 0; c < C_; ++c)
        acc += siluf(temb[b * C_ + c]) * W[n * C_ + c];
    (sel ? te : tn)[b * C_ + n] = acc + Bi[n];
}

// h0 = (x + tn[b]) * node_mask
__global__ __launch_bounds__(256) void h0_kernel(
    const float* __restrict__ x, const float* __restrict__ tn,
    const float* __restrict__ nm, float* __restrict__ h0)
{
    int m = blockIdx.x, c = threadIdx.x;
    int b = m / N_;
    h0[(long)m * C_ + c] = (x[(long)m * C_ + c] + tn[b * C_ + c]) * nm[m];
}

// GINE gather: t1[dst] += relu(h0[src] + (de[b,i,j]+te[b])*adj[b,i,j])
__global__ __launch_bounds__(256) void gine_gather_kernel(
    const float* __restrict__ h0, const float* __restrict__ de,
    const float* __restrict__ te, const float* __restrict__ adj,
    const int* __restrict__ src, const int* __restrict__ dst,
    const int* __restrict__ eb, const int* __restrict__ ei,
    const int* __restrict__ ej, float* __restrict__ t1)
{
    int e = blockIdx.x, c = threadIdx.x;
    int b = eb[e], i = ei[e], j = ej[e];
    long m = ((long)(b * N_ + i)) * N_ + j;
    float v = h0[(long)src[e] * C_ + c] +
              (de[m * C_ + c] + te[b * C_ + c]) * adj[m];
    atomicAdd(&t1[(long)dst[e] * C_ + c], fmaxf(v, 0.f));
}

// Row GroupNorm: groups of C/G=8 channels -> lane L owns group L exactly.
__global__ __launch_bounds__(32) void gn_rows_kernel(
    float* __restrict__ out, const float* __restrict__ a,
    const float* __restrict__ res, const float* __restrict__ nm,
    const float* __restrict__ gamma, const float* __restrict__ beta,
    int premask, int postmask)
{
    int m = blockIdx.x, lane = threadIdx.x;
    float mask = nm[m];
    float x[8], mu = 0.f;
    #pragma unroll
    for (int t = 0; t < 8; ++t) {
        int c = lane * 8 + t;
        float v = a[(long)m * C_ + c];
        if (premask) v *= mask;
        if (res) v += res[(long)m * C_ + c];
        x[t] = v; mu += v;
    }
    mu *= 0.125f;
    float var = 0.f;
    #pragma unroll
    for (int t = 0; t < 8; ++t) { float d = x[t] - mu; var += d * d; }
    var *= 0.125f;
    float rr = rsqrtf(var + EPS_);
    #pragma unroll
    for (int t = 0; t < 8; ++t) {
        int c = lane * 8 + t;
        float y = (x[t] - mu) * rr * gamma[c] + beta[c];
        if (postmask) y *= mask;
        out[(long)m * C_ + c] = y;
    }
}

// h1 = (h_local + h_attn) * nm
__global__ __launch_bounds__(256) void combine_kernel(
    const float* __restrict__ hl, const float* __restrict__ ha,
    const float* __restrict__ nm, float* __restrict__ h1)
{
    int m = blockIdx.x, c = threadIdx.x;
    h1[(long)m * C_ + c] = (hl[(long)m * C_ + c] + ha[(long)m * C_ + c]) * nm[m];
}

// Edge-gated full attention for one (b, dst=j): segment softmax over src i.
__global__ __launch_bounds__(256) void attn_kernel(
    const float* __restrict__ q, const float* __restrict__ k,
    const float* __restrict__ v, const float* __restrict__ e0,
    const float* __restrict__ e1, float* __restrict__ out)
{
    const int bj = blockIdx.x;
    const int b = bj / N_, j = bj % N_;
    __shared__ float qsh[C_];
    __shared__ float alpha[H_][N_];
    const int tid = threadIdx.x;

    qsh[tid] = q[(long)bj * C_ + tid];
    __syncthreads();

    for (int task = tid; task < H_ * N_; task += 256) {
        int hh = task / N_, i = task - hh * N_;
        const float* kr = &k[((long)(b * N_ + i)) * C_ + hh * HC_];
        const float* er = &e0[(((long)(b * N_) + i) * N_ + j) * C_ + hh * HC_];
        float a = 0.f;
        #pragma unroll 8
        for (int c = 0; c < HC_; ++c) a += qsh[hh * HC_ + c] * kr[c] * er[c];
        alpha[hh][i] = (i == j) ? -1e30f : a * 0.17677669529663687f;
    }
    __syncthreads();

    int wave = tid >> 5, lane = tid & 31;
    float vals[3], mx = -1e30f;
    #pragma unroll
    for (int t = 0; t < 3; ++t) { vals[t] = alpha[wave][lane + 32 * t]; mx = fmaxf(mx, vals[t]); }
    for (int off = 16; off; off >>= 1) mx = fmaxf(mx, __shfl_xor(mx, off, 32));
    float s = 0.f;
    #pragma unroll
    for (int t = 0; t < 3; ++t) { vals[t] = __expf(vals[t] - mx); s += vals[t]; }
    for (int off = 16; off; off >>= 1) s += __shfl_xor(s, off, 32);
    float inv = 1.f / (s + 1e-16f);
    #pragma unroll
    for (int t = 0; t < 3; ++t) alpha[wave][lane + 32 * t] = vals[t] * inv;
    __syncthreads();

    int hh = tid >> 5, c = tid & 31;
    float acc = 0.f;
    for (int i = 0; i < N_; ++i) {
        float w = alpha[hh][i];
        acc += w * v[((long)(b * N_ + i)) * C_ + hh * HC_ + c] *
                   e1[(((long)(b * N_) + i) * N_ + j) * C_ + hh * HC_ + c];
    }
    out[(long)bj * C_ + tid] = acc;
}

// Edge GroupNorm pass 1: stats per (b, g) over (N,N,8)
__global__ __launch_bounds__(256) void gn_edge_stats_kernel(
    const float* __restrict__ x, float* __restrict__ stats)
{
    int bg = blockIdx.x;
    int b = bg >> 5, g = bg & 31;
    __shared__ float ssum[256], ssq[256];
    float s = 0.f, sq = 0.f;
    for (int pos = threadIdx.x; pos < N_ * N_; pos += 256) {
        const float* p = x + ((long)(b * N_ * N_ + pos)) * C_ + g * 8;
        #pragma unroll
        for (int cc = 0; cc < 8; ++cc) { float v = p[cc]; s += v; sq += v * v; }
    }
    ssum[threadIdx.x] = s; ssq[threadIdx.x] = sq;
    __syncthreads();
    for (int off = 128; off; off >>= 1) {
        if (threadIdx.x < off) {
            ssum[threadIdx.x] += ssum[threadIdx.x + off];
            ssq[threadIdx.x]  += ssq[threadIdx.x + off];
        }
        __syncthreads();
    }
    if (threadIdx.x == 0) {
        float cnt = (float)(N_ * N_ * 8);
        float mu = ssum[0] / cnt;
        stats[bg * 2 + 0] = mu;
        stats[bg * 2 + 1] = ssq[0] / cnt - mu * mu;
    }
}

// Edge GroupNorm pass 2 (in place on d_out edge region) + adj mask
__global__ __launch_bounds__(256) void gn_edge_norm_kernel(
    float* __restrict__ x, const float* __restrict__ stats,
    const float* __restrict__ gamma, const float* __restrict__ beta,
    const float* __restrict__ adj)
{
    long m = blockIdx.x;
    int c = threadIdx.x;
    int b = (int)(m / (N_ * N_));
    int g = c >> 3;
    float mu = stats[(b * G_ + g) * 2 + 0];
    float var = stats[(b * G_ + g) * 2 + 1];
    float v = x[m * C_ + c];
    x[m * C_ + c] = ((v - mu) * rsqrtf(var + EPS_) * gamma[c] + beta[c]) * adj[m];
}

// ===========================================================================
// Host side
// ===========================================================================
static inline void launch_gemm(hipStream_t s, const float* A, const __bf16* Abf,
                               const __bf16* Wbf, const float* bias,
                               const float* res, float* out, __bf16* outbf,
                               int M, int K, int Nout, int mode, int act,
                               const float* te, const float* adj, const float* hd)
{
    dim3 grid(M / BM, Nout / BN);
    switch (mode) {
    case 0: gemm_wmma_kernel<0><<<grid, 128, 0, s>>>(A, Abf, Wbf, bias, res, out, outbf,
                                                     K, Nout, act, te, adj, hd); break;
    case 1: gemm_wmma_kernel<1><<<grid, 128, 0, s>>>(A, Abf, Wbf, bias, res, out, outbf,
                                                     K, Nout, act, te, adj, hd); break;
    case 2: gemm_wmma_kernel<2><<<grid, 128, 0, s>>>(A, Abf, Wbf, bias, res, out, outbf,
                                                     K, Nout, act, te, adj, hd); break;
    default: gemm_wmma_kernel<3><<<grid, 128, 0, s>>>(A, Abf, Wbf, bias, res, out, outbf,
                                                      K, Nout, act, te, adj, hd); break;
    }
}

static inline void cvt_w(hipStream_t s, const float* src, __bf16* dst, int n)
{
    cvt_bf16_kernel<<<(n + 255) / 256, 256, 0, s>>>(src, dst, n);
}

extern "C" void kernel_launch(void* const* d_in, const int* in_sizes, int n_in,
                              void* d_out, int out_size, void* d_ws, size_t ws_size,
                              hipStream_t stream)
{
    (void)n_in; (void)out_size; (void)ws_size;
    const float* x    = (const float*)d_in[0];
    const float* de   = (const float*)d_in[1];
    const float* temb = (const float*)d_in[2];
    const float* nm   = (const float*)d_in[3];
    const float* adj  = (const float*)d_in[4];
    const int* loc_src = (const int*)d_in[5];
    const int* loc_dst = (const int*)d_in[6];
    const int* loc_b   = (const int*)d_in[7];
    const int* loc_i   = (const int*)d_in[8];
    const int* loc_j   = (const int*)d_in[9];
    const int E_loc = in_sizes[5];
    const float* t_node_w = (const float*)d_in[15];
    const float* t_node_b = (const float*)d_in[16];
    const float* t_edge_w = (const float*)d_in[17];
    const float* t_edge_b = (const float*)d_in[18];
    const float* gin_w1 = (const float*)d_in[19];
    const float* gin_b1 = (const float*)d_in[20];
    const float* gin_w2 = (const float*)d_in[21];
    const float* gin_b2 = (const float*)d_in[22];
    const float* q_w = (const float*)d_in[23];
    const float* q_b = (const float*)d_in[24];
    const float* k_w = (const float*)d_in[25];
    const float* k_b = (const float*)d_in[26];
    const float* v_w = (const float*)d_in[27];
    const float* v_b = (const float*)d_in[28];
    const float* e0_w = (const float*)d_in[29];
    const float* e1_w = (const float*)d_in[30];
    const float* gn1l_g = (const float*)d_in[31];
    const float* gn1l_b = (const float*)d_in[32];
    const float* gn1a_g = (const float*)d_in[33];
    const float* gn1a_b = (const float*)d_in[34];
    const float* ff1_w = (const float*)d_in[35];
    const float* ff1_b = (const float*)d_in[36];
    const float* ff2_w = (const float*)d_in[37];
    const float* ff2_b = (const float*)d_in[38];
    const float* gn2n_g = (const float*)d_in[39];
    const float* gn2n_b = (const float*)d_in[40];
    const float* ff3_w = (const float*)d_in[41];
    const float* ff3_b = (const float*)d_in[42];
    const float* ff4_w = (const float*)d_in[43];
    const float* ff4_b = (const float*)d_in[44];
    const float* gn2e_g = (const float*)d_in[45];
    const float* gn2e_b = (const float*)d_in[46];

    const long NODE_E = (long)NODES_ * C_;
    const long EDGE_E = (long)EDGEROWS_ * C_;
    const int CC  = C_ * C_;          // 65536
    const int CC2 = 2 * C_ * C_;      // 131072

    float* ws = (float*)d_ws;
    float* tn = ws;
    float* te = ws + 2048;
    float* h0 = ws + 4096;
    float* t1 = h0 + NODE_E;
    float* t2 = t1 + NODE_E;
    float* gin = t2 + NODE_E;
    float* hloc = gin + NODE_E;
    float* qb = hloc + NODE_E;
    float* kb = qb + NODE_E;
    float* vb = kb + NODE_E;
    float* attno = vb + NODE_E;
    float* hattn = attno + NODE_E;
    float* h1 = hattn + NODE_E;
    float* h2 = h1 + NODE_E;
    float* t3 = h2 + NODE_E;
    float* stats = t3 + (long)NODES_ * 512;
    // bf16 weight pool (11 matrices, 983040 bf16 = 491520 floats)
    __bf16* wbf = (__bf16*)(stats + 512);
    __bf16* wb_gin1 = wbf;
    __bf16* wb_gin2 = wb_gin1 + CC;
    __bf16* wb_q    = wb_gin2 + CC;
    __bf16* wb_k    = wb_q + CC;
    __bf16* wb_v    = wb_k + CC;
    __bf16* wb_e0   = wb_v + CC;
    __bf16* wb_e1   = wb_e0 + CC;
    __bf16* wb_ff1  = wb_e1 + CC;
    __bf16* wb_ff2  = wb_ff1 + CC2;
    __bf16* wb_ff3  = wb_ff2 + CC2;
    __bf16* wb_ff4  = wb_ff3 + CC2;
    float* big = (float*)(wb_ff4 + CC2);
    float* e0 = big;
    float* e1 = big + EDGE_E;
    __bf16* u1bf = (__bf16*)big;       // [73728,512] bf16; reuses e0/e1 region

    float* out_h  = (float*)d_out;
    float* out_he = (float*)d_out + NODE_E;

    // ---- weight conversions (f32 -> bf16, L2 resident) ----
    cvt_w(stream, gin_w1, wb_gin1, CC);
    cvt_w(stream, gin_w2, wb_gin2, CC);
    cvt_w(stream, q_w, wb_q, CC);
    cvt_w(stream, k_w, wb_k, CC);
    cvt_w(stream, v_w, wb_v, CC);
    cvt_w(stream, e0_w, wb_e0, CC);
    cvt_w(stream, e1_w, wb_e1, CC);
    cvt_w(stream, ff1_w, wb_ff1, CC2);
    cvt_w(stream, ff2_w, wb_ff2, CC2);
    cvt_w(stream, ff3_w, wb_ff3, CC2);
    cvt_w(stream, ff4_w, wb_ff4, CC2);

    temb_proj_kernel<<<16, 256, 0, stream>>>(temb, t_node_w, t_node_b,
                                             t_edge_w, t_edge_b, tn, te);
    h0_kernel<<<NODES_, 256, 0, stream>>>(x, tn, nm, h0);
    hipMemcpyAsync(t1, h0, NODE_E * sizeof(float), hipMemcpyDeviceToDevice, stream);
    gine_gather_kernel<<<E_loc, 256, 0, stream>>>(h0, de, te, adj, loc_src,
                                                  loc_dst, loc_b, loc_i, loc_j, t1);
    launch_gemm(stream, t1, nullptr, wb_gin1, gin_b1, nullptr, t2, nullptr,
                NODES_, C_, C_, 0, 1, nullptr, nullptr, nullptr);
    launch_gemm(stream, t2, nullptr, wb_gin2, gin_b2, nullptr, gin, nullptr,
                NODES_, C_, C_, 0, 0, nullptr, nullptr, nullptr);
    gn_rows_kernel<<<NODES_, 32, 0, stream>>>(hloc, gin, x, nm, gn1l_g, gn1l_b, 1, 0);
    launch_gemm(stream, h0, nullptr, wb_q, q_b, nullptr, qb, nullptr,
                NODES_, C_, C_, 0, 0, nullptr, nullptr, nullptr);
    launch_gemm(stream, h0, nullptr, wb_k, k_b, nullptr, kb, nullptr,
                NODES_, C_, C_, 0, 0, nullptr, nullptr, nullptr);
    launch_gemm(stream, h0, nullptr, wb_v, v_b, nullptr, vb, nullptr,
                NODES_, C_, C_, 0, 0, nullptr, nullptr, nullptr);
    launch_gemm(stream, de, nullptr, wb_e0, nullptr, nullptr, e0, nullptr,
                EDGEROWS_, C_, C_, 1, 0, te, adj, nullptr);
    launch_gemm(stream, de, nullptr, wb_e1, nullptr, nullptr, e1, nullptr,
                EDGEROWS_, C_, C_, 1, 3, te, adj, nullptr);
    attn_kernel<<<NODES_, 256, 0, stream>>>(qb, kb, vb, e0, e1, attno);
    gn_rows_kernel<<<NODES_, 32, 0, stream>>>(hattn, attno, x, nm, gn1a_g, gn1a_b, 0, 0);
    combine_kernel<<<NODES_, 256, 0, stream>>>(hloc, hattn, nm, h1);
    launch_gemm(stream, h1, nullptr, wb_ff1, ff1_b, nullptr, t3, nullptr,
                NODES_, C_, 2 * C_, 0, 2, nullptr, nullptr, nullptr);
    launch_gemm(stream, t3, nullptr, wb_ff2, ff2_b, h1, h2, nullptr,
                NODES_, 2 * C_, C_, 0, 0, nullptr, nullptr, nullptr);
    gn_rows_kernel<<<NODES_, 32, 0, stream>>>(out_h, h2, nullptr, nm, gn2n_g, gn2n_b, 0, 1);
    // edge FF: u1 (bf16) = silu(h_edge2 @ ff3^T + b3); he = de + u1 @ ff4^T + b4
    launch_gemm(stream, nullptr, nullptr, wb_ff3, ff3_b, nullptr, nullptr, u1bf,
                EDGEROWS_, C_, 2 * C_, 2, 2, nullptr, nullptr, h1);
    launch_gemm(stream, nullptr, u1bf, wb_ff4, ff4_b, de, out_he, nullptr,
                EDGEROWS_, 2 * C_, C_, 3, 0, nullptr, nullptr, nullptr);
    gn_edge_stats_kernel<<<B_ * G_, 256, 0, stream>>>(out_he, stats);
    gn_edge_norm_kernel<<<EDGEROWS_, 256, 0, stream>>>(out_he, stats, gn2e_g, gn2e_b, adj);
}
